// Attention_LoRA_14018773254461
// MI455X (gfx1250) — compile-verified, matrix operand order checked by reading, not run on hardware
//
#include <hip/hip_runtime.h>
#include <hip/hip_bf16.h>

// ---------------- Problem constants ----------------
#define BATCH 64
#define SEQ   197
#define CDIM  768
#define HEADS 12
#define HDIM  64
#define RANK  24
#define THREE_C 2304
#define MROWS (BATCH * SEQ)          // 12608
#define QSCALE 0.125f                // 64^-0.5
#define NPAD 224                     // 14 tiles of 16 key rows (padded)

typedef __attribute__((ext_vector_type(16))) __bf16 v16bf;
typedef __attribute__((ext_vector_type(8)))  __bf16 v8bf;
typedef __attribute__((ext_vector_type(8)))  float  v8f;
typedef __attribute__((ext_vector_type(4)))  unsigned int u32x4;
typedef __attribute__((ext_vector_type(8)))  int i32x8;
typedef __attribute__((ext_vector_type(4)))  int i32x4;

// ---------------- Tensor Data Mover: 2D tile load, bf16 (data_size=2B) ----------
// Descriptor packing per CDNA5 ISA 8.3/8.4. OOB rows (y >= tensor_d1) are
// zero-filled in LDS; pad_interval/pad_amount insert LDS row padding
// (codes: interval 2<<c DWORDs, amount c+1 DWORDs).
__device__ inline void tdm_load_2d(const void* lds_ptr, const void* gptr,
                                   unsigned tensor_d0, unsigned tensor_d1,
                                   unsigned tile_d0, unsigned tile_d1,
                                   unsigned long long stride_elems,
                                   unsigned pad_int_code, unsigned pad_amt_code) {
    unsigned long long ga = (unsigned long long)(uintptr_t)gptr;
    u32x4 g0;
    g0[0] = 1u;                                        // count=1 (valid), user mode
    g0[1] = (unsigned)(uintptr_t)lds_ptr;              // lds_addr (byte offset)
    g0[2] = (unsigned)ga;                              // global_addr[31:0]
    g0[3] = (unsigned)((ga >> 32) & 0x01FFFFFFu) | (2u << 30);  // [56:32] | type=2
    i32x8 g1;
    unsigned w0 = (1u << 16)                           // data_size = 2 bytes
                | (1u << 20)                           // pad_enable
                | (pad_int_code << 22) | (pad_amt_code << 25);
    g1[0] = (int)w0;
    g1[1] = (int)((tensor_d0 & 0xFFFFu) << 16);        // tensor_dim0[15:0] @ [63:48]
    g1[2] = (int)((tensor_d0 >> 16) | ((tensor_d1 & 0xFFFFu) << 16));
    g1[3] = (int)((tensor_d1 >> 16) | (tile_d0 << 16)); // tile_dim0 @ [127:112]
    g1[4] = (int)(tile_d1 & 0xFFFFu);                  // tile_dim1; tile_dim2=0
    g1[5] = (int)(stride_elems & 0xFFFFFFFFu);         // tensor_dim0_stride[31:0]
    g1[6] = (int)((stride_elems >> 32) & 0xFFFFu);     // stride[47:32]; dim1_stride=0
    g1[7] = 0;
    i32x4 gz = {0, 0, 0, 0};
#if defined(__clang_major__) && __clang_major__ >= 23
    i32x8 gz8 = {0, 0, 0, 0, 0, 0, 0, 0};
    __builtin_amdgcn_tensor_load_to_lds(g0, g1, gz, gz, gz8, 0);
#else
    __builtin_amdgcn_tensor_load_to_lds(g0, g1, gz, gz, 0);
#endif
}

// ---------------- WMMA fragment helpers ----------------
// Per-lane layout for v_wmma_*_16x16x32_bf16 (ISA 7.12.2):
//   lanes 0-15 : row = lane,    k in {0..7, 16..23}
//   lanes 16-31: row = lane-16, k in {8..15, 24..31}
__device__ inline v16bf ldsFrag(const __bf16* base, int strideElems) {
    const int lane = threadIdx.x & 31;
    const int row  = lane & 15;
    const int k8   = (lane >> 4) << 3;
    const __bf16* p = base + row * strideElems + k8;
    v16bf f;
    ((v8bf*)&f)[0] = *(const v8bf*)(p);
    ((v8bf*)&f)[1] = *(const v8bf*)(p + 16);
    return f;
}

__device__ inline v16bf gFragGuard(const __bf16* rowBase, int row, int rowLimit,
                                   int strideElems, int kofs) {
    const int lane = threadIdx.x & 31;
    const int r    = row + (lane & 15);
    const int k8   = (lane >> 4) << 3;
    v16bf f;
    if (r < rowLimit) {
        const __bf16* p = rowBase + (size_t)r * strideElems + kofs + k8;
        ((v8bf*)&f)[0] = *(const v8bf*)(p);
        ((v8bf*)&f)[1] = *(const v8bf*)(p + 16);
    } else {
        for (int i = 0; i < 16; ++i) f[i] = (__bf16)0.0f;
    }
    return f;
}

__device__ inline v8f wmma_bf16(v16bf a, v16bf b, v8f c) {
    return __builtin_amdgcn_wmma_f32_16x16x32_bf16(false, a, false, b, (short)0, c,
                                                   false, false);
}

// ---------------- Small prep kernels ----------------
__global__ void cvt_f32_to_bf16(const float* __restrict__ in,
                                __bf16* __restrict__ out, int n) {
    int i = blockIdx.x * 256 + threadIdx.x;
    if (i < n) out[i] = (__bf16)in[i];
}

// W_eff[d, c] = qkv_w[d, c] + lora_b[d%768, r] * lora_a[r, c]  (rank 24, per section)
__global__ void build_weff(const float* __restrict__ qkv_w,
                           const float* __restrict__ qa, const float* __restrict__ qb,
                           const float* __restrict__ ka, const float* __restrict__ kb,
                           const float* __restrict__ va, const float* __restrict__ vb,
                           __bf16* __restrict__ weff) {
    int i = blockIdx.x * 256 + threadIdx.x;
    if (i >= THREE_C * CDIM) return;
    int d = i / CDIM, c = i - d * CDIM;
    int sec = d / CDIM;           // 0=q,1=k,2=v
    int dr  = d - sec * CDIM;
    const float* A; const float* Bm;
    if (sec == 0)      { A = qa; Bm = qb; }
    else if (sec == 1) { A = ka; Bm = kb; }
    else               { A = va; Bm = vb; }
    float w = qkv_w[i];
    for (int r = 0; r < RANK; ++r)
        w += Bm[dr * RANK + r] * A[r * CDIM + c];
    weff[i] = (__bf16)w;
}

// ---------------- Tiled bf16 WMMA GEMM with TDM double-buffered staging -------
// C[m, j] = sum_k A[m,k] * Bw[j,k]   (multiply by Bw^T; Bw row-major [N,K])
// mode 0: + qkv bias, q scaled by QSCALE, bf16 scatter -> qkv_out[which][b][h][n][d]
// mode 1: + proj_b, fp32 store -> out_f32[m*768 + j]
#define GM 128
#define GN 128
#define GK 32
#define LSTR 40   // padded LDS row stride: 32 elems + 4-DWORD TDM pad

__global__ __launch_bounds__(256) void gemm_bf16(
    const __bf16* __restrict__ A, const __bf16* __restrict__ Bw,
    int M, int Ncols, int K, int mode,
    const float* __restrict__ qb, const float* __restrict__ vbias,
    __bf16* __restrict__ qkv_out,
    const float* __restrict__ pb, float* __restrict__ out_f32) {
    __shared__ __bf16 tiles[2 * (GM + GN) * LSTR];     // double-buffered A|B tiles
    const int tid  = threadIdx.x;
    const int wave = tid >> 5;
    const int lane = tid & 31;
    const int bm   = blockIdx.x * GM;
    const int bn   = blockIdx.y * GN;
    const int wm   = (wave & 3) * 32;   // 4 waves along M
    const int wn   = (wave >> 2) * 64;  // 2 waves along N
    const float zf = 0.0f;
    v8f zero = {zf, zf, zf, zf, zf, zf, zf, zf};
    v8f acc[2][4];
    for (int i = 0; i < 2; ++i)
        for (int j = 0; j < 4; ++j) acc[i][j] = zero;

    const int nk = K / GK;
    // pad codes: interval 16 DWORDs (code 3) = 32 bf16 row, amount 4 DWORDs (code 3)
    if (wave == 0) {
        tdm_load_2d(tiles, A + (size_t)bm * K,
                    (unsigned)K, (unsigned)(M - bm), GK, GM, (unsigned)K, 3, 3);
        tdm_load_2d(tiles + GM * LSTR, Bw + (size_t)bn * K,
                    (unsigned)K, (unsigned)(Ncols - bn), GK, GN, (unsigned)K, 3, 3);
    }
    for (int i = 0; i < nk; ++i) {
        const __bf16* Acur = tiles + (i & 1) * (GM + GN) * LSTR;
        const __bf16* Bcur = Acur + GM * LSTR;
        if (wave == 0) {
            if (i + 1 < nk) {
                const int kt = (i + 1) * GK;
                __bf16* Anx = tiles + ((i + 1) & 1) * (GM + GN) * LSTR;
                tdm_load_2d(Anx, A + (size_t)bm * K + kt,
                            (unsigned)(K - kt), (unsigned)(M - bm), GK, GM,
                            (unsigned)K, 3, 3);
                tdm_load_2d(Anx + GM * LSTR, Bw + (size_t)bn * K + kt,
                            (unsigned)(K - kt), (unsigned)(Ncols - bn), GK, GN,
                            (unsigned)K, 3, 3);
                __builtin_amdgcn_s_wait_tensorcnt(2);  // stage i resident (in-order)
            } else {
                __builtin_amdgcn_s_wait_tensorcnt(0);
            }
        }
        __syncthreads();
        v16bf af[2], bfrag[4];
        af[0] = ldsFrag(Acur + (wm +  0) * LSTR, LSTR);
        af[1] = ldsFrag(Acur + (wm + 16) * LSTR, LSTR);
        for (int j = 0; j < 4; ++j)
            bfrag[j] = ldsFrag(Bcur + (wn + j * 16) * LSTR, LSTR);
        for (int ii = 0; ii < 2; ++ii)
            for (int j = 0; j < 4; ++j)
                acc[ii][j] = wmma_bf16(af[ii], bfrag[j], acc[ii][j]);
        __syncthreads();
    }

    const int lrow = lane & 15;
    const int lhi8 = (lane >> 4) << 3;
    if (mode == 0) {
        for (int i = 0; i < 2; ++i) {
            const int rb = bm + wm + i * 16 + lhi8;
            int bbv[8], nnv[8];
#pragma unroll
            for (int vi = 0; vi < 8; ++vi) {
                int m = rb + vi;
                int bb = m / SEQ;
                bbv[vi] = bb; nnv[vi] = m - bb * SEQ;
            }
            for (int j = 0; j < 4; ++j) {
                int col   = bn + wn + j * 16 + lrow;
                int which = col / CDIM;
                int cc    = col - which * CDIM;
                int hh    = cc >> 6, dd = cc & 63;
                float bias = (which == 0) ? qb[cc] : (which == 2 ? vbias[cc] : 0.0f);
                float scl  = (which == 0) ? QSCALE : 1.0f;
#pragma unroll
                for (int vi = 0; vi < 8; ++vi) {
                    int m = rb + vi;
                    if (m < M) {
                        size_t dst = ((((size_t)which * BATCH + bbv[vi]) * HEADS + hh)
                                          * SEQ + nnv[vi]) * HDIM + dd;
                        qkv_out[dst] = (__bf16)((acc[i][j][vi] + bias) * scl);
                    }
                }
            }
        }
    } else {
        for (int i = 0; i < 2; ++i) {
            const int rb = bm + wm + i * 16 + lhi8;
            for (int j = 0; j < 4; ++j) {
                int col = bn + wn + j * 16 + lrow;
                float bias = pb[col];
#pragma unroll
                for (int vi = 0; vi < 8; ++vi) {
                    int m = rb + vi;
                    if (m < M) out_f32[(size_t)m * CDIM + col] = acc[i][j][vi] + bias;
                }
            }
        }
    }
}

// ---------------- Attention kernel ----------------
// One workgroup (4 waves) per (b,h). K staged via TDM (row pad -> stride 72,
// OOB rows 197..223 zero-filled), V^T staged manually. Per-wave 16-row query
// tiles: S = Q K^T (WMMA) + rel-pos bias, fp32 softmax through per-wave LDS
// with shfl_xor(16) cross-half reduction, then P.V (WMMA).
#define KSTR 72     // 64 elems + 4-DWORD TDM pad
#define VSTR 232    // V^T / P LDS row stride (bf16)
#define SSTR 225    // S LDS row stride (f32)

__global__ __launch_bounds__(128) void attn_kernel(
    const __bf16* __restrict__ qkvb, const float* __restrict__ table,
    const int* __restrict__ ridx, __bf16* __restrict__ attnb) {
    extern __shared__ char smem[];
    __bf16* Ks  = (__bf16*)smem;                         // 224*72  bf16
    __bf16* VTs = Ks + NPAD * KSTR;                      // 64*232  bf16
    float*  Sb  = (float*)(VTs + HDIM * VSTR);           // 4*16*225 f32
    __bf16* Pb  = (__bf16*)(Sb + 4 * 16 * SSTR);         // 4*16*232 bf16

    const int bh = blockIdx.x;
    const int b  = bh / HEADS, h = bh - b * HEADS;
    const size_t qbase = (((size_t)(0 * BATCH + b) * HEADS + h) * SEQ) * HDIM;
    const size_t kbase = (((size_t)(1 * BATCH + b) * HEADS + h) * SEQ) * HDIM;
    const size_t vbase = (((size_t)(2 * BATCH + b) * HEADS + h) * SEQ) * HDIM;

    const int tid  = threadIdx.x;
    const int wave = tid >> 5, lane = tid & 31;

    // K: TDM 2D tile 64x224, tensor rows=197 -> tail zero-filled, pad -> KSTR.
    if (wave == 0)
        tdm_load_2d(Ks, qkvb + kbase, HDIM, SEQ, HDIM, NPAD, HDIM, 4, 3);
    // V^T: manual transpose while the TDM runs.
    for (int it = 0; it < 14; ++it) {
        int e = (it * 128 + tid) * 8;        // 0..14328
        int r = e >> 6, c = e & 63;
        v8bf vv;
        if (r < SEQ) vv = *(const v8bf*)(qkvb + vbase + r * HDIM + c);
        else         for (int q = 0; q < 8; ++q) vv[q] = (__bf16)0.0f;
        for (int q = 0; q < 8; ++q) VTs[(c + q) * VSTR + r] = vv[q];
    }
    if (wave == 0) __builtin_amdgcn_s_wait_tensorcnt(0);
    __syncthreads();

    const int lrow = lane & 15, lhi = lane >> 4;
    float*  Sw = Sb + wave * 16 * SSTR;
    __bf16* Pw = Pb + wave * 16 * VSTR;

    for (int t = wave; t < 13; t += 4) {      // 13 query tiles of 16 rows
        const int tbase = t * 16;
        v16bf aq0 = gFragGuard(qkvb + qbase, tbase, SEQ, HDIM, 0);
        v16bf aq1 = gFragGuard(qkvb + qbase, tbase, SEQ, HDIM, 32);

        // ---- scores ----
        for (int j = 0; j < 14; ++j) {
            const float zf = 0.0f;
            v8f acc = {zf, zf, zf, zf, zf, zf, zf, zf};
            v16bf kb0 = ldsFrag(Ks + j * 16 * KSTR +  0, KSTR);
            v16bf kb1 = ldsFrag(Ks + j * 16 * KSTR + 32, KSTR);
            acc = wmma_bf16(aq0, kb0, acc);
            acc = wmma_bf16(aq1, kb1, acc);
            const int m = j * 16 + lrow;
            for (int vi = 0; vi < 8; ++vi) {
                int rt = vi + lhi * 8;
                int n  = tbase + rt;
                float s;
                if (m < SEQ)
                    s = (n < SEQ) ? acc[vi] + table[ridx[n * SEQ + m] * HEADS + h] : 0.0f;
                else
                    s = -1e30f;
                Sw[rt * SSTR + m] = s;
            }
        }

        // ---- softmax (lane l: row l&15, half l>>4 of 224 cols) ----
        const int srow = lrow;
        const int c0   = lhi * 112;
        float mx = -3.0e38f;
        for (int c = 0; c < 112; ++c) mx = fmaxf(mx, Sw[srow * SSTR + c0 + c]);
        mx = fmaxf(mx, __shfl_xor(mx, 16, 32));
        float sum = 0.0f;
        for (int c = 0; c < 112; ++c) {
            float e = __expf(Sw[srow * SSTR + c0 + c] - mx);
            sum += e;
            Sw[srow * SSTR + c0 + c] = e;
        }
        sum += __shfl_xor(sum, 16, 32);
        float rinv = 1.0f / sum;
        for (int c = 0; c < 112; ++c)
            Pw[srow * VSTR + c0 + c] = (__bf16)(Sw[srow * SSTR + c0 + c] * rinv);

        // ---- out = P @ V ----
        for (int tn = 0; tn < 4; ++tn) {
            const float zf = 0.0f;
            v8f o = {zf, zf, zf, zf, zf, zf, zf, zf};
            for (int kc = 0; kc < 7; ++kc) {
                v16bf pa = ldsFrag(Pw + kc * 32, VSTR);
                v16bf vb = ldsFrag(VTs + (tn * 16) * VSTR + kc * 32, VSTR);
                o = wmma_bf16(pa, vb, o);
            }
            const int dcol = tn * 16 + lrow;
            for (int vi = 0; vi < 8; ++vi) {
                int n = tbase + vi + lhi * 8;
                if (n < SEQ)
                    attnb[((size_t)b * SEQ + n) * CDIM + h * HDIM + dcol] = (__bf16)o[vi];
            }
        }
    }
}

// ---------------- Launcher ----------------
extern "C" void kernel_launch(void* const* d_in, const int* in_sizes, int n_in,
                              void* d_out, int out_size, void* d_ws, size_t ws_size,
                              hipStream_t stream) {
    const float* x        = (const float*)d_in[0];
    const float* qkv_w    = (const float*)d_in[1];
    const float* q_bias   = (const float*)d_in[2];
    const float* v_bias   = (const float*)d_in[3];
    const float* q_lora_a = (const float*)d_in[4];
    const float* q_lora_b = (const float*)d_in[5];
    const float* k_lora_a = (const float*)d_in[6];
    const float* k_lora_b = (const float*)d_in[7];
    const float* v_lora_a = (const float*)d_in[8];
    const float* v_lora_b = (const float*)d_in[9];
    const float* rel_tab  = (const float*)d_in[10];
    const float* proj_w   = (const float*)d_in[11];
    const float* proj_b   = (const float*)d_in[12];
    const int*   rel_idx  = (const int*)d_in[13];
    float* out = (float*)d_out;

    // workspace carve (all offsets 256B aligned)
    char* ws = (char*)d_ws;
    __bf16* xb   = (__bf16*)(ws);                              // 12608*768   bf16
    __bf16* weff = (__bf16*)(ws + 19365888);                   // 2304*768    bf16
    __bf16* pwb  = (__bf16*)(ws + 19365888 + 3538944);         // 768*768     bf16
    __bf16* qkvb = (__bf16*)(ws + 19365888 + 3538944 + 1179648);            // 3*B*H*N*64
    __bf16* attb = (__bf16*)(ws + 19365888 + 3538944 + 1179648 + 58097664); // 12608*768

    // 1) precision conversion + LoRA fold
    {
        int n = MROWS * CDIM;
        cvt_f32_to_bf16<<<(n + 255) / 256, 256, 0, stream>>>(x, xb, n);
        int nw = CDIM * CDIM;
        cvt_f32_to_bf16<<<(nw + 255) / 256, 256, 0, stream>>>(proj_w, pwb, nw);
        int ne = THREE_C * CDIM;
        build_weff<<<(ne + 255) / 256, 256, 0, stream>>>(
            qkv_w, q_lora_a, q_lora_b, k_lora_a, k_lora_b, v_lora_a, v_lora_b, weff);
    }
    // 2) fused QKV GEMM  (M=12608, N=2304, K=768)
    {
        dim3 grid((MROWS + GM - 1) / GM, THREE_C / GN);
        gemm_bf16<<<grid, 256, 0, stream>>>(xb, weff, MROWS, THREE_C, CDIM, /*mode=*/0,
                                            q_bias, v_bias, qkvb, nullptr, nullptr);
    }
    // 3) attention per (b,h)
    {
        size_t smem = (size_t)NPAD * KSTR * 2 + (size_t)HDIM * VSTR * 2 +
                      (size_t)4 * 16 * SSTR * 4 + (size_t)4 * 16 * VSTR * 2; // 149248 B
        attn_kernel<<<BATCH * HEADS, 128, smem, stream>>>(qkvb, rel_tab, rel_idx, attb);
    }
    // 4) output projection (M=12608, N=768, K=768) -> fp32 d_out
    {
        dim3 grid((MROWS + GM - 1) / GM, CDIM / GN);
        gemm_bf16<<<grid, 256, 0, stream>>>(attb, pwb, MROWS, CDIM, CDIM, /*mode=*/1,
                                            nullptr, nullptr, nullptr, proj_b, out);
    }
}
